// Aggregation_89575837925422
// MI455X (gfx1250) — matrix operand order repaired and checked by reading, not compiled, not run on hardware
//
#include <hip/hip_runtime.h>
#include <hip/hip_bf16.h>

// ---------------------------------------------------------------------------
// Problem constants (from the reference)
// ---------------------------------------------------------------------------
#define BB    2
#define CC    768        // CIN
#define HH    44
#define WW    44
#define NP    1936       // H*W
#define NT    121        // NP / 16
#define NHD   4          // heads
#define DH    192        // head dim
#define CO    384        // conv out channels
#define CF    1920       // feats channels = 768 + 3*384
#define MAXP  100

typedef __attribute__((ext_vector_type(16))) __bf16        v16bf;
typedef __attribute__((ext_vector_type(8)))  float         v8f;
typedef __attribute__((ext_vector_type(4)))  unsigned int  u32x4;
typedef __attribute__((ext_vector_type(8)))  int           i32x8;
typedef __attribute__((ext_vector_type(4)))  int           i32x4;

static __device__ __forceinline__ v8f wmma_bf16(v16bf a, v16bf b, v8f c) {
    return __builtin_amdgcn_wmma_f32_16x16x32_bf16(false, a, false, b,
                                                   (short)0, c, false, false);
}

// A fragment (16x32 bf16): element (M,K): M = lane&15,
// idx 0..7  -> K = kbase + idx        (kbase = 8 for lanes 16..31)
// idx 8..15 -> K = kbase + 16 + idx-8
static __device__ __forceinline__ v16bf load_a(const __bf16* base, int stride, int lane) {
    const __bf16* p = base + (size_t)(lane & 15) * stride + ((lane >> 4) << 3);
    v16bf a;
#pragma unroll
    for (int i = 0; i < 8; ++i) { a[i] = p[i]; a[8 + i] = p[16 + i]; }
    return a;
}

// A fragment with per-lane row pointer; nullptr row -> zeros (conv padding)
static __device__ __forceinline__ v16bf load_a_row(const __bf16* rowp, int lane) {
    v16bf a;
#pragma unroll
    for (int i = 0; i < 16; ++i) a[i] = (__bf16)0.0f;
    if (rowp) {
        const __bf16* p = rowp + ((lane >> 4) << 3);
#pragma unroll
        for (int i = 0; i < 8; ++i) { a[i] = p[i]; a[8 + i] = p[16 + i]; }
    }
    return a;
}

// B fragment (32x16 bf16): element (K,N): N = lane&15,
// idx 0..7 -> K = kbase + idx, idx 8..15 -> K = kbase + 16 + idx-8
// (kbase = 8 for lanes 16..31).  Memory: one row per N, contiguous along K.
static __device__ __forceinline__ v16bf load_b(const __bf16* base, int stride, int lane) {
    const __bf16* p = base + (size_t)(lane & 15) * stride + ((lane >> 4) << 3);
    v16bf b;
#pragma unroll
    for (int i = 0; i < 8; ++i) { b[i] = p[i]; b[8 + i] = p[16 + i]; }
    return b;
}

// ---------------------------------------------------------------------------
// Tensor Data Mover: 2D tile load (global -> LDS), data_size = 2 bytes.
// tensor_d0/d1 are the extents *remaining from the tile start* so that the
// TDM's OOB rule (reads beyond tensor dim return zero) zero-fills tails.
// D# packing per CDNA5 ISA 8.3/8.4 (group0: count/lds/global/type=2,
// group1: data_size, dims, tile dims, dim0 stride).
// amdgpu-toolchain (clang-23) 6-arg builtin form.
// ---------------------------------------------------------------------------
static __device__ __forceinline__ void tdm_load_2d(unsigned lds_addr,
                                                   const void* gptr,
                                                   unsigned tile_d0, unsigned tile_d1,
                                                   unsigned long long tensor_d0,
                                                   unsigned long long tensor_d1,
                                                   unsigned long long stride0) {
    const unsigned long long ga = (unsigned long long)gptr;
    u32x4 g0;
    g0[0] = 1u;                                            // count=1 (valid)
    g0[1] = lds_addr;                                      // LDS byte address
    g0[2] = (unsigned)(ga & 0xffffffffull);                // global addr lo
    g0[3] = (unsigned)((ga >> 32) & 0x1ffffffull) | (2u << 30);  // hi | type=2

    i32x8 g1;
    g1[0] = (1 << 16);                                     // data_size=1 -> 2B
    g1[1] = (int)((tensor_d0 & 0xffffull) << 16);          // td0[15:0]
    g1[2] = (int)(((tensor_d0 >> 16) & 0xffffull) |
                  ((tensor_d1 & 0xffffull) << 16));        // td0[31:16] | td1[15:0]
    g1[3] = (int)(((tensor_d1 >> 16) & 0xffffull) |
                  ((unsigned long long)tile_d0 << 16));    // td1[31:16] | tile_d0
    g1[4] = (int)(tile_d1 & 0xffffu);                      // tile_d1 | tile_d2=0
    g1[5] = (int)(stride0 & 0xffffffffull);                // dim0 stride lo
    g1[6] = (int)((stride0 >> 32) & 0xffffull);            // stride hi | s1 lo=0
    g1[7] = 0;

    i32x4 z4; z4[0] = 0; z4[1] = 0; z4[2] = 0; z4[3] = 0;  // 2D: groups 2/3 unused
    i32x8 z8;
#pragma unroll
    for (int i = 0; i < 8; ++i) z8[i] = 0;
    __builtin_amdgcn_tensor_load_to_lds(g0, g1, z4, z4, z8, 0);
}

// ---------------------------------------------------------------------------
// Packing kernels (fp32 -> bf16, layout changes)
// ---------------------------------------------------------------------------
__global__ void k_pack_x(const float* __restrict__ x, __bf16* __restrict__ feats) {
    const int total = BB * NP * CC;
    for (int idx = blockIdx.x * blockDim.x + threadIdx.x; idx < total;
         idx += gridDim.x * blockDim.x) {
        int c = idx % CC;
        int p = (idx / CC) % NP;
        int b = idx / (CC * NP);
        feats[((size_t)(b * NP + p)) * CF + c] =
            (__bf16)x[((size_t)(b * CC + c)) * NP + p];
    }
}

// w (CO, CC, 3, 3) -> wt[tap][o][c]
__global__ void k_pack_convw(const float* __restrict__ w, __bf16* __restrict__ wt) {
    const int total = 9 * CO * CC;
    for (int idx = blockIdx.x * blockDim.x + threadIdx.x; idx < total;
         idx += gridDim.x * blockDim.x) {
        int c   = idx % CC;
        int o   = (idx / CC) % CO;
        int tap = idx / (CC * CO);
        wt[(size_t)tap * CO * CC + (size_t)o * CC + c] =
            (__bf16)w[((size_t)o * CC + c) * 9 + tap];
    }
}

__global__ void k_pack_mat(const float* __restrict__ src, __bf16* __restrict__ dst, int n) {
    for (int idx = blockIdx.x * blockDim.x + threadIdx.x; idx < n;
         idx += gridDim.x * blockDim.x)
        dst[idx] = (__bf16)src[idx];
}

// ---------------------------------------------------------------------------
// Dilated 3x3 convs as 9 per-tap WMMA GEMMs (double-buffered K loop).
// One wave: 16 pixels x 64 outputs.  grid = (121, 3*6, B)
// ---------------------------------------------------------------------------
__global__ __launch_bounds__(32) void k_conv(__bf16* __restrict__ feats,
                                             const __bf16* __restrict__ wt_all,
                                             const float* __restrict__ b3,
                                             const float* __restrict__ b5,
                                             const float* __restrict__ b7) {
    const int lane = threadIdx.x;
    const int pt = blockIdx.x;
    const int di = blockIdx.y / 6, og = blockIdx.y % 6;
    const int b  = blockIdx.z;
    const int dil = (di == 0) ? 3 : (di == 1) ? 5 : 7;
    const float* bias = (di == 0) ? b3 : (di == 1) ? b5 : b7;
    const __bf16* wt = wt_all + (size_t)di * 9 * CO * CC;

    const int p  = pt * 16 + (lane & 15);
    const int py = p / WW, px = p % WW;

    v8f acc[4];
#pragma unroll
    for (int t = 0; t < 4; ++t)
#pragma unroll
        for (int r = 0; r < 8; ++r) acc[t][r] = 0.0f;

    for (int tap = 0; tap < 9; ++tap) {
        const int dy = (tap / 3 - 1) * dil, dx = (tap % 3 - 1) * dil;
        const int sy = py + dy, sx = px + dx;
        const bool ok = (sy >= 0) && (sy < HH) && (sx >= 0) && (sx < WW);
        const __bf16* arow =
            ok ? feats + ((size_t)(b * NP + sy * WW + sx)) * CF : (const __bf16*)0;
        const __bf16* wtap = wt + (size_t)tap * CO * CC;

        v16bf aC = load_a_row(arow, lane);
        v16bf bC[4];
#pragma unroll
        for (int t = 0; t < 4; ++t)
            bC[t] = load_b(wtap + (size_t)(og * 64 + t * 16) * CC, CC, lane);

        for (int kk = 0; kk < CC / 32; ++kk) {
            const int kn = (kk < CC / 32 - 1) ? kk + 1 : kk;
            v16bf aN = load_a_row(ok ? arow + kn * 32 : (const __bf16*)0, lane);
            v16bf bN[4];
#pragma unroll
            for (int t = 0; t < 4; ++t)
                bN[t] = load_b(wtap + (size_t)(og * 64 + t * 16) * CC + kn * 32, CC, lane);
#pragma unroll
            for (int t = 0; t < 4; ++t) acc[t] = wmma_bf16(aC, bC[t], acc[t]);
            aC = aN;
#pragma unroll
            for (int t = 0; t < 4; ++t) bC[t] = bN[t];
        }
    }

    const int n_ = lane & 15, half = lane >> 4;
#pragma unroll
    for (int t = 0; t < 4; ++t) {
        const int o = og * 64 + t * 16 + n_;
        const float bv = bias[o];
#pragma unroll
        for (int r = 0; r < 8; ++r) {
            const int pp = pt * 16 + r + 8 * half;
            feats[((size_t)(b * NP + pp)) * CF + CC + di * CO + o] =
                (__bf16)(acc[t][r] + bv);
        }
    }
}

// ---------------------------------------------------------------------------
// Generic 16x64 WMMA GEMM body used by qk / v / proj (double-buffered)
// ---------------------------------------------------------------------------
static __device__ __forceinline__ void gemm_16x64(const __bf16* arow0, int astride,
                                                  const __bf16* brow0, int bstride,
                                                  int ksteps, int lane, v8f acc[4]) {
    v16bf aC = load_a(arow0, astride, lane);
    v16bf bC[4];
#pragma unroll
    for (int t = 0; t < 4; ++t)
        bC[t] = load_b(brow0 + (size_t)(t * 16) * bstride, bstride, lane);

    for (int kk = 0; kk < ksteps; ++kk) {
        const int kn = (kk < ksteps - 1) ? kk + 1 : kk;
        v16bf aN = load_a(arow0 + kn * 32, astride, lane);
        v16bf bN[4];
#pragma unroll
        for (int t = 0; t < 4; ++t)
            bN[t] = load_b(brow0 + (size_t)(t * 16) * bstride + kn * 32, bstride, lane);
#pragma unroll
        for (int t = 0; t < 4; ++t) acc[t] = wmma_bf16(aC, bC[t], acc[t]);
        aC = aN;
#pragma unroll
        for (int t = 0; t < 4; ++t) bC[t] = bN[t];
    }
}

// qk GEMM: feats (N,1920) @ w_qk^T -> qh/kh (B,H,N,192) bf16.  grid=(121,24,B)
__global__ __launch_bounds__(32) void k_qk(const __bf16* __restrict__ feats,
                                           const __bf16* __restrict__ wqk,
                                           __bf16* __restrict__ qh,
                                           __bf16* __restrict__ kh) {
    const int lane = threadIdx.x;
    const int pt = blockIdx.x, og = blockIdx.y, b = blockIdx.z;

    v8f acc[4];
#pragma unroll
    for (int t = 0; t < 4; ++t)
#pragma unroll
        for (int r = 0; r < 8; ++r) acc[t][r] = 0.0f;

    gemm_16x64(feats + (size_t)(b * NP + pt * 16) * CF, CF,
               wqk + (size_t)(og * 64) * CF, CF, CF / 32, lane, acc);

    const int n_ = lane & 15, half = lane >> 4;
#pragma unroll
    for (int t = 0; t < 4; ++t) {
        const int o = og * 64 + t * 16 + n_;
        const bool isq = o < CC;
        const int oo = isq ? o : o - CC;
        const int hh = oo / DH, dc = oo % DH;
        const float sc = isq ? 0.5f : 1.0f;   // HEAD^-0.5 folded into q
        __bf16* dst = isq ? qh : kh;
#pragma unroll
        for (int r = 0; r < 8; ++r) {
            const int pp = pt * 16 + r + 8 * half;
            dst[((size_t)((b * NHD + hh) * NP + pp)) * DH + dc] = (__bf16)(acc[t][r] * sc);
        }
    }
}

// v GEMM: x (N,768) @ w_v^T -> vT (B,H,192,N) bf16.  grid=(121,12,B)
__global__ __launch_bounds__(32) void k_v(const __bf16* __restrict__ feats,
                                          const __bf16* __restrict__ wv,
                                          __bf16* __restrict__ vT) {
    const int lane = threadIdx.x;
    const int pt = blockIdx.x, og = blockIdx.y, b = blockIdx.z;

    v8f acc[4];
#pragma unroll
    for (int t = 0; t < 4; ++t)
#pragma unroll
        for (int r = 0; r < 8; ++r) acc[t][r] = 0.0f;

    gemm_16x64(feats + (size_t)(b * NP + pt * 16) * CF, CF,
               wv + (size_t)(og * 64) * CC, CC, CC / 32, lane, acc);

    const int n_ = lane & 15, half = lane >> 4;
#pragma unroll
    for (int t = 0; t < 4; ++t) {
        const int o = og * 64 + t * 16 + n_;
        const int hh = o / DH, dc = o % DH;
#pragma unroll
        for (int r = 0; r < 8; ++r) {
            const int pp = pt * 16 + r + 8 * half;
            vT[((size_t)((b * NHD + hh) * DH + dc)) * NP + pp] = (__bf16)(acc[t][r]);
        }
    }
}

// ---------------------------------------------------------------------------
// Relative-position bias tables
// ---------------------------------------------------------------------------
__global__ void k_qr(const __bf16* __restrict__ qh,
                     const float* __restrict__ rel_h, const float* __restrict__ rel_w,
                     float* __restrict__ qrh, float* __restrict__ qrw) {
    const int total = BB * NHD * NP * HH;
    for (int idx = blockIdx.x * blockDim.x + threadIdx.x; idx < total;
         idx += gridDim.x * blockDim.x) {
        const int u  = idx % HH;
        const int p  = (idx / HH) % NP;
        const int bh = idx / (HH * NP);
        const int x_ = p / WW, y_ = p % WW;
        const __bf16* qrow = qh + ((size_t)bh * NP + p) * DH;
        const float* rh = rel_h + (size_t)(u - x_ + MAXP - 1) * DH;
        const float* rw = rel_w + (size_t)(u - y_ + MAXP - 1) * DH;
        float s1 = 0.0f, s2 = 0.0f;
        for (int d = 0; d < DH; ++d) {
            const float qd = (float)qrow[d];
            s1 += qd * rh[d];
            s2 += qd * rw[d];
        }
        qrh[idx] = s1;
        qrw[idx] = s2;
    }
}

// ---------------------------------------------------------------------------
// Flash attention with TDM double-buffered K/V staging.
// One wave per (b, h, 16-query tile).  Per pair of key tiles (32 keys):
//   K tile 32x192 and V slab 192x32 are DMA'd to LDS by the Tensor Data Mover
//   (TENSORcnt, double buffered), S = q@k^T via WMMA from LDS B-frags, bias,
//   online softmax, P staged via LDS -> A-frag, O += P@V via WMMA.
// grid = (121, 4, B)
// ---------------------------------------------------------------------------
__global__ __launch_bounds__(32) void k_attn(const __bf16* __restrict__ qh,
                                             const __bf16* __restrict__ kh,
                                             const __bf16* __restrict__ vT,
                                             const float* __restrict__ qrh,
                                             const float* __restrict__ qrw,
                                             __bf16* __restrict__ outT) {
    __shared__ __bf16 kbuf[2 * 32 * DH];    // [buf][key(32)][d(192)]
    __shared__ __bf16 vbuf[2 * DH * 32];    // [buf][d(192)][key(32)]
    __shared__ __bf16 lds_p[16 * 32];

    const int lane = threadIdx.x;
    const int qt = blockIdx.x, h = blockIdx.y, b = blockIdx.z;
    const int bh = b * NHD + h;
    const size_t bhN = (size_t)bh * NP;
    const int n_ = lane & 15, half = lane >> 4;

    const __bf16* khb = kh + bhN * DH;
    const __bf16* vtb = vT + (size_t)bh * DH * NP;
    const float* qrh_b = qrh + bhN * HH;
    const float* qrw_b = qrw + bhN * HH;

    // q fragments for the whole K=192 strip
    v16bf qf[6];
    const __bf16* qbase = qh + (bhN + qt * 16) * DH;
#pragma unroll
    for (int kb = 0; kb < 6; ++kb) qf[kb] = load_a(qbase + kb * 32, DH, lane);

    v8f O[12];
#pragma unroll
    for (int t = 0; t < 12; ++t)
#pragma unroll
        for (int r = 0; r < 8; ++r) O[t][r] = 0.0f;

    float mrow[8], lrow[8];
#pragma unroll
    for (int r = 0; r < 8; ++r) { mrow[r] = -3.0e38f; lrow[r] = 0.0f; }

    // TDM issue for key-pair `pair` into buffer `buf`
    auto issue = [&](int pair, int buf) {
        // K tile: 32 keys x 192 d  (row stride 192); rows past NP -> zeros
        tdm_load_2d((unsigned)(size_t)&kbuf[buf * 32 * DH],
                    khb + (size_t)(32 * pair) * DH,
                    DH, 32,
                    (unsigned long long)DH,
                    (unsigned long long)(NP - 32 * pair),
                    (unsigned long long)DH);
        // V slab: 192 d rows x 32 keys (row stride NP); cols past NP -> zeros
        tdm_load_2d((unsigned)(size_t)&vbuf[buf * DH * 32],
                    vtb + 32 * pair,
                    32, DH,
                    (unsigned long long)(NP - 32 * pair),
                    (unsigned long long)DH,
                    (unsigned long long)NP);
    };

    issue(0, 0);

    for (int i = 0; i < 61; ++i) {          // 61 pairs; last half-pair phantom
        if (i < 60) {
            issue(i + 1, (i + 1) & 1);
            __builtin_amdgcn_s_wait_tensorcnt(2);   // current buffer ready
        } else {
            __builtin_amdgcn_s_wait_tensorcnt(0);
        }
        const __bf16* kc = kbuf + (i & 1) * 32 * DH;
        const __bf16* vc = vbuf + (i & 1) * DH * 32;

        float S[2][8];
#pragma unroll
        for (int s = 0; s < 2; ++s) {
            const int jt = 2 * i + s;
            if (jt < NT) {
                v8f Sv;
#pragma unroll
                for (int r = 0; r < 8; ++r) Sv[r] = 0.0f;
#pragma unroll
                for (int kb = 0; kb < 6; ++kb) {
                    v16bf bf_ = load_b(kc + (size_t)(s * 16) * DH + kb * 32, DH, lane);
                    Sv = wmma_bf16(qf[kb], bf_, Sv);
                }
                const int kp = jt * 16 + n_;
                const int u = kp / WW, vv = kp % WW;
#pragma unroll
                for (int r = 0; r < 8; ++r) {
                    const int qp = qt * 16 + r + 8 * half;
                    S[s][r] = Sv[r] + qrh_b[(size_t)qp * HH + u] + qrw_b[(size_t)qp * HH + vv];
                }
            } else {
#pragma unroll
                for (int r = 0; r < 8; ++r) S[s][r] = -1.0e30f;
            }
        }

        // row max over 32 key columns
        float tm[8];
#pragma unroll
        for (int r = 0; r < 8; ++r) tm[r] = fmaxf(S[0][r], S[1][r]);
#pragma unroll
        for (int m = 1; m <= 8; m <<= 1)
#pragma unroll
            for (int r = 0; r < 8; ++r) tm[r] = fmaxf(tm[r], __shfl_xor(tm[r], m, 32));

        float osc[8];
#pragma unroll
        for (int r = 0; r < 8; ++r) {
            const float mn = fmaxf(mrow[r], tm[r]);
            osc[r] = __expf(mrow[r] - mn);
            mrow[r] = mn;
        }

        float P0[8], P1[8], ts[8];
#pragma unroll
        for (int r = 0; r < 8; ++r) {
            P0[r] = __expf(S[0][r] - mrow[r]);
            P1[r] = __expf(S[1][r] - mrow[r]);
            ts[r] = P0[r] + P1[r];
        }
#pragma unroll
        for (int m = 1; m <= 8; m <<= 1)
#pragma unroll
            for (int r = 0; r < 8; ++r) ts[r] += __shfl_xor(ts[r], m, 32);
#pragma unroll
        for (int r = 0; r < 8; ++r) lrow[r] = lrow[r] * osc[r] + ts[r];
#pragma unroll
        for (int t = 0; t < 12; ++t)
#pragma unroll
            for (int r = 0; r < 8; ++r) O[t][r] *= osc[r];

        // C-layout P -> row-major 16x32 bf16 in LDS -> A fragment
        __syncthreads();
#pragma unroll
        for (int r = 0; r < 8; ++r) {
            const int row = r + 8 * half;
            lds_p[row * 32 + n_]      = (__bf16)P0[r];
            lds_p[row * 32 + 16 + n_] = (__bf16)P1[r];
        }
        __syncthreads();
        v16bf pa = load_a(lds_p, 32, lane);

#pragma unroll
        for (int t = 0; t < 12; ++t) {
            v16bf vb = load_b(vc + (size_t)(t * 16) * 32, 32, lane);
            O[t] = wmma_bf16(pa, vb, O[t]);
        }
    }

    float inv[8];
#pragma unroll
    for (int r = 0; r < 8; ++r) inv[r] = 1.0f / lrow[r];
#pragma unroll
    for (int t = 0; t < 12; ++t)
#pragma unroll
        for (int r = 0; r < 8; ++r) {
            const int pp = qt * 16 + r + 8 * half;
            const int c = h * DH + t * 16 + n_;
            outT[((size_t)(b * NP + pp)) * CC + c] = (__bf16)(O[t][r] * inv[r]);
        }
}

// ---------------------------------------------------------------------------
// Projection + residual: out = x + gamma * (outT @ w_proj^T).  grid=(121,12,B)
// ---------------------------------------------------------------------------
__global__ __launch_bounds__(32) void k_proj(const __bf16* __restrict__ outT,
                                             const __bf16* __restrict__ wproj,
                                             const float* __restrict__ x,
                                             const float* __restrict__ gamma,
                                             float* __restrict__ out) {
    const int lane = threadIdx.x;
    const int pt = blockIdx.x, og = blockIdx.y, b = blockIdx.z;

    v8f acc[4];
#pragma unroll
    for (int t = 0; t < 4; ++t)
#pragma unroll
        for (int r = 0; r < 8; ++r) acc[t][r] = 0.0f;

    gemm_16x64(outT + (size_t)(b * NP + pt * 16) * CC, CC,
               wproj + (size_t)(og * 64) * CC, CC, CC / 32, lane, acc);

    const float g = gamma[0];
    const int n_ = lane & 15, half = lane >> 4;
#pragma unroll
    for (int t = 0; t < 4; ++t) {
        const int o = og * 64 + t * 16 + n_;
#pragma unroll
        for (int r = 0; r < 8; ++r) {
            const int pp = pt * 16 + r + 8 * half;
            const size_t oi = ((size_t)(b * CC + o)) * NP + pp;
            out[oi] = x[oi] + g * acc[t][r];
        }
    }
}

// ---------------------------------------------------------------------------
// Host launch
// ---------------------------------------------------------------------------
extern "C" void kernel_launch(void* const* d_in, const int* in_sizes, int n_in,
                              void* d_out, int out_size, void* d_ws, size_t ws_size,
                              hipStream_t stream) {
    const float* x      = (const float*)d_in[0];
    const float* w_d3   = (const float*)d_in[1];
    const float* b_d3   = (const float*)d_in[2];
    const float* w_d5   = (const float*)d_in[3];
    const float* b_d5   = (const float*)d_in[4];
    const float* w_d7   = (const float*)d_in[5];
    const float* b_d7   = (const float*)d_in[6];
    const float* w_qk   = (const float*)d_in[7];
    const float* rel_h  = (const float*)d_in[8];
    const float* rel_w  = (const float*)d_in[9];
    const float* w_v    = (const float*)d_in[10];
    const float* w_proj = (const float*)d_in[11];
    const float* gamma  = (const float*)d_in[12];
    float* out = (float*)d_out;

    char* ws = (char*)d_ws;
    size_t off = 0;
    auto alloc = [&](size_t bytes) -> void* {
        void* p = ws + off;
        off = (off + bytes + 255) & ~(size_t)255;
        return p;
    };

    __bf16* feats = (__bf16*)alloc((size_t)BB * NP * CF * 2);
    __bf16* wt    = (__bf16*)alloc((size_t)3 * 9 * CO * CC * 2);
    __bf16* wqkb  = (__bf16*)alloc((size_t)2 * CC * CF * 2);
    __bf16* wvb   = (__bf16*)alloc((size_t)CC * CC * 2);
    __bf16* wpb   = (__bf16*)alloc((size_t)CC * CC * 2);
    __bf16* qh    = (__bf16*)alloc((size_t)BB * NHD * NP * DH * 2);
    __bf16* kh    = (__bf16*)alloc((size_t)BB * NHD * NP * DH * 2);
    __bf16* vT    = (__bf16*)alloc((size_t)BB * NHD * DH * NP * 2);
    float*  qrh   = (float*)alloc((size_t)BB * NHD * NP * HH * 4);
    float*  qrw   = (float*)alloc((size_t)BB * NHD * NP * HH * 4);
    __bf16* outT  = (__bf16*)alloc((size_t)BB * NP * CC * 2);
    (void)ws_size; (void)in_sizes; (void)n_in; (void)out_size;

    {
        int n = BB * NP * CC;
        k_pack_x<<<(n + 255) / 256, 256, 0, stream>>>(x, feats);
    }
    {
        int n = 9 * CO * CC;
        k_pack_convw<<<(n + 255) / 256, 256, 0, stream>>>(w_d3, wt + (size_t)0 * 9 * CO * CC);
        k_pack_convw<<<(n + 255) / 256, 256, 0, stream>>>(w_d5, wt + (size_t)1 * 9 * CO * CC);
        k_pack_convw<<<(n + 255) / 256, 256, 0, stream>>>(w_d7, wt + (size_t)2 * 9 * CO * CC);
    }
    {
        int n = 2 * CC * CF;
        k_pack_mat<<<(n + 255) / 256, 256, 0, stream>>>(w_qk, wqkb, n);
        n = CC * CC;
        k_pack_mat<<<(n + 255) / 256, 256, 0, stream>>>(w_v, wvb, n);
        k_pack_mat<<<(n + 255) / 256, 256, 0, stream>>>(w_proj, wpb, n);
    }

    k_conv<<<dim3(NT, 18, BB), 32, 0, stream>>>(feats, wt, b_d3, b_d5, b_d7);
    k_qk  <<<dim3(NT, 24, BB), 32, 0, stream>>>(feats, wqkb, qh, kh);
    k_v   <<<dim3(NT, 12, BB), 32, 0, stream>>>(feats, wvb, vT);
    {
        int n = BB * NHD * NP * HH;
        k_qr<<<(n + 255) / 256, 256, 0, stream>>>(qh, rel_h, rel_w, qrh, qrw);
    }
    k_attn<<<dim3(NT, NHD, BB), 32, 0, stream>>>(qh, kh, vT, qrh, qrw, outT);
    k_proj<<<dim3(NT, 12, BB), 32, 0, stream>>>(outT, wpb, x, gamma, out);
}
